// MultiInterestLayer_32950989095092
// MI455X (gfx1250) — compile-verified, matrix operand order checked by reading, not run on hardware
//
#include <hip/hip_runtime.h>
#include <hip/hip_bf16.h>
#include <stdint.h>

typedef __attribute__((ext_vector_type(16))) __bf16 v16bf;
typedef __attribute__((ext_vector_type(8)))  float  v8f;

#define B_SZ   1024
#define L_SZ   200
#define L_PAD  224
#define D_SZ   256
#define K_SZ   8
#define NEG_INF_F (-4294967295.0f)

// workspace layout (bytes)
#define WS_MAPPED_OFF   0ull                       // bf16 [B][L][D] = 104,857,600 B
#define WS_WFRAG_OFF    104857600ull               // u32  [8][16][32][8] = 131,072 B
#define WS_LOGITS_OFF   (104857600ull + 131072ull) // f32 [K][L] (+pad)
#define WS_DELTA_OFF    (WS_LOGITS_OFF + 8192ull)  // f32 [B][K][L] = 6,553,600 B

// k-offset of the first element of u32-pair i within a 32-wide K chunk,
// per the CDNA5 16-bit A/B fragment layout (lanes 0-15 hold K {0..7,16..23},
// lanes 16-31 hold K {8..15,24..31}).
__device__ __forceinline__ int koff_pair(int i, int half) {
  return ((i < 4) ? 0 : 16) + 8 * half + 2 * (i & 3);
}

__device__ __forceinline__ unsigned short f2bf(float f) {
  return __builtin_bit_cast(unsigned short, (__bf16)f);
}

// ---------------------------------------------------------------------------
// CDNA5 async global<->LDS DMA (ASYNCcnt-tracked, bypasses VGPRs).
// global_load_async_to_lds_b128: VDST = LDS byte address, VADDR = 64-bit addr.
// ---------------------------------------------------------------------------
__device__ __forceinline__ void async_load_lds_b128(unsigned lds_addr,
                                                    const void* gaddr) {
  asm volatile("global_load_async_to_lds_b128 %0, %1, off"
               :
               : "v"(lds_addr), "v"((unsigned long long)(uintptr_t)gaddr)
               : "memory");
}

__device__ __forceinline__ void async_store_lds_b128(void* gaddr,
                                                     unsigned lds_addr) {
  asm volatile("global_store_async_from_lds_b128 %0, %1, off"
               :
               : "v"((unsigned long long)(uintptr_t)gaddr), "v"(lds_addr)
               : "memory");
}

__device__ __forceinline__ void wait_asynccnt0() {
  asm volatile("s_wait_asynccnt 0" ::: "memory");
}

__device__ __forceinline__ unsigned lds_addr_of(const void* p) {
  // generic pointers to LDS carry the LDS byte offset in the low 32 bits
  return (unsigned)(uintptr_t)p;
}

// ---------------------------------------------------------------------------
// Pre-swizzle W (f32 [256][256]) into bf16 B-matrix fragment order:
// wfrag[kk][n][lane][i] packs (W[k0][col], W[k0+1][col]) for the WMMA lane map.
// ---------------------------------------------------------------------------
__global__ void prep_wfrag(const float* __restrict__ W, unsigned* __restrict__ wfrag) {
  int tid  = blockIdx.x * 256 + threadIdx.x;   // 0..32767
  int i    = tid & 7;
  int lane = (tid >> 3) & 31;
  int n    = (tid >> 8) & 15;
  int kk   = tid >> 12;
  int half = lane >> 4;
  int col  = n * 16 + (lane & 15);
  int k0   = kk * 32 + koff_pair(i, half);
  unsigned lo = f2bf(W[k0 * D_SZ + col]);
  unsigned hi = f2bf(W[(k0 + 1) * D_SZ + col]);
  wfrag[tid] = lo | (hi << 16);
}

__global__ void init_logits(const float* __restrict__ rl, float* __restrict__ logits) {
  int i = blockIdx.x * 256 + threadIdx.x;
  if (i < K_SZ * L_SZ) logits[i] = rl[i];
}

// ---------------------------------------------------------------------------
// mapped = X @ W  (bf16 inputs, f32 accumulate, bf16 output to workspace)
// block = 256 threads = 8 waves; each wave computes a 16x256 strip (128 rows/block)
// W fragments staged per block in LDS via async DMA (128 KB dynamic); output
// repacked through LDS and streamed out with async b128 stores.
// ---------------------------------------------------------------------------
__global__ void __launch_bounds__(256, 1) gemm_map(
    const float* __restrict__ X, const unsigned* __restrict__ wfrag,
    unsigned short* __restrict__ mapped) {
  extern __shared__ char smem[];
  unsigned* wl = (unsigned*)smem;  // 32768 u32 = 128 KB

  // async DMA: W fragments -> LDS (8192 x b128), no VGPR round-trip
  {
    const uint4* src = (const uint4*)wfrag;
    for (int i = threadIdx.x; i < 8192; i += 256)
      async_load_lds_b128(lds_addr_of(&wl[i * 4]), src + i);
  }
  wait_asynccnt0();
  __syncthreads();

  const int wave = threadIdx.x >> 5;
  const int lane = threadIdx.x & 31;
  const int lr = lane & 15, lh = lane >> 4;
  const long rowBase = (long)blockIdx.x * 128 + wave * 16;
  const float* xrow = X + (rowBase + lr) * D_SZ;

  const v8f zero = {0.f, 0.f, 0.f, 0.f, 0.f, 0.f, 0.f, 0.f};
  v8f acc[16];
#pragma unroll
  for (int n = 0; n < 16; ++n) acc[n] = zero;

  for (int kk = 0; kk < 8; ++kk) {
    const float4* p = (const float4*)(xrow + kk * 32 + 8 * lh);
    float4 f0 = p[0];
    float4 f1 = p[1];
    const float4* q = (const float4*)(xrow + kk * 32 + 16 + 8 * lh);
    float4 f2 = q[0];
    float4 f3 = q[1];

    v16bf a;
    a[0]  = (__bf16)f0.x; a[1]  = (__bf16)f0.y; a[2]  = (__bf16)f0.z; a[3]  = (__bf16)f0.w;
    a[4]  = (__bf16)f1.x; a[5]  = (__bf16)f1.y; a[6]  = (__bf16)f1.z; a[7]  = (__bf16)f1.w;
    a[8]  = (__bf16)f2.x; a[9]  = (__bf16)f2.y; a[10] = (__bf16)f2.z; a[11] = (__bf16)f2.w;
    a[12] = (__bf16)f3.x; a[13] = (__bf16)f3.y; a[14] = (__bf16)f3.z; a[15] = (__bf16)f3.w;

    const v16bf* bfrags = (const v16bf*)(wl + (size_t)kk * 16 * 32 * 8);
#pragma unroll
    for (int n = 0; n < 16; ++n) {
      v16bf bf = bfrags[n * 32 + lane];
      acc[n] = __builtin_amdgcn_wmma_f32_16x16x32_bf16(
          false, a, false, bf, (short)0, acc[n], false, false);
    }
  }

  // repack C fragments (bf16) through LDS, reusing the W-fragment space
  __syncthreads();
  unsigned short* ot = (unsigned short*)smem;  // [128][256] bf16 = 64 KB
#pragma unroll
  for (int n = 0; n < 16; ++n) {
#pragma unroll
    for (int r = 0; r < 8; ++r) {
      int lrow = wave * 16 + r + 8 * lh;   // C-matrix M = r (+8 for upper lanes)
      int col = n * 16 + lr;
      ot[lrow * D_SZ + col] = f2bf(acc[n][r]);
    }
  }
  __syncthreads();

  // stream the 64 KB tile to global, coalesced b128 straight from LDS
  {
    uint4* dst = (uint4*)(mapped + (size_t)blockIdx.x * 128 * D_SZ);
    for (int i = threadIdx.x; i < 4096; i += 256)
      async_store_lds_b128(dst + i, lds_addr_of(&ot[i * 8]));
  }
  wait_asynccnt0();
}

// ---------------------------------------------------------------------------
// One routing iteration for batch b = blockIdx.x:
//  - async-DMA mapped[b] (bf16, L padded to 224 rows) into LDS
//  - masked softmax over L per capsule (wave-per-k, shfl reductions)
//  - Z = weight @ mapped via WMMA (M=16 pad, 7 K-chunks, 2 N-tiles/wave)
//  - squash via ds-atomic norm accumulation
//  - caps written to d_out and LDS
//  - delta_part[b] = caps @ mapped^T via WMMA (8 K-chunks, 13 L-tiles)
// ---------------------------------------------------------------------------
__global__ void __launch_bounds__(256, 1) routing(
    const unsigned short* __restrict__ mapped, const int* __restrict__ seq_len,
    const float* __restrict__ logits, float* __restrict__ out,
    float* __restrict__ delta_part) {
  extern __shared__ char smem[];
  __bf16* m    = (__bf16*)smem;                             // [224][256] bf16
  float* wlds  = (float*)(smem + (size_t)L_PAD * D_SZ * 2); // [8][224]
  float* caps  = wlds + K_SZ * L_PAD;                       // [8][256]
  float* norm  = caps + K_SZ * D_SZ;                        // [8]
  float* scale = norm + K_SZ;                               // [8]

  const int b = blockIdx.x;
  const int tid = threadIdx.x;
  const int wave = tid >> 5, lane = tid & 31, lr = lane & 15, lh = lane >> 4;

  // async DMA: mapped[b] -> LDS (6400 x b128); zero the L padding rows
  {
    const uint4* src = (const uint4*)(mapped + (size_t)b * L_SZ * D_SZ);
    unsigned* dst = (unsigned*)m;
    for (int i = tid; i < (L_SZ * D_SZ) / 8; i += 256)
      async_load_lds_b128(lds_addr_of(&dst[i * 4]), src + i);
    for (int i = (L_SZ * D_SZ) / 2 + tid; i < (L_PAD * D_SZ) / 2; i += 256)
      dst[i] = 0u;
  }
  const int len = seq_len[b];
  for (int i = tid; i < K_SZ * L_PAD; i += 256) {
    int k = i / L_PAD, l = i - k * L_PAD;
    wlds[i] = (l < len) ? logits[k * L_SZ + l] : NEG_INF_F;
  }
  if (tid < K_SZ) norm[tid] = 0.0f;
  wait_asynccnt0();
  __syncthreads();

  // softmax: wave w owns capsule k = w
  {
    const int k = wave;
    float mx = NEG_INF_F;
    for (int l = lane; l < L_PAD; l += 32) mx = fmaxf(mx, wlds[k * L_PAD + l]);
#pragma unroll
    for (int s = 16; s > 0; s >>= 1) mx = fmaxf(mx, __shfl_xor(mx, s, 32));
    float sum = 0.f;
    for (int l = lane; l < L_PAD; l += 32) {
      float e = __expf(wlds[k * L_PAD + l] - mx);  // masked entries -> 0
      wlds[k * L_PAD + l] = e;
      sum += e;
    }
#pragma unroll
    for (int s = 16; s > 0; s >>= 1) sum += __shfl_xor(sum, s, 32);
    float inv = 1.0f / sum;
    for (int l = lane; l < L_PAD; l += 32) wlds[k * L_PAD + l] *= inv;
  }
  __syncthreads();

  const v8f zero = {0.f, 0.f, 0.f, 0.f, 0.f, 0.f, 0.f, 0.f};

  // Z = weight(16xL_PAD, rows>=8 zero) @ mapped(L_PAD x 256)
  v8f z0 = zero, z1 = zero;
  const int n0 = wave * 2, n1 = wave * 2 + 1;
  for (int kk = 0; kk < 7; ++kk) {
    v16bf a;
#pragma unroll
    for (int i = 0; i < 8; ++i) {
      int k0 = kk * 32 + koff_pair(i, lh);
      float x0 = (lr < K_SZ) ? wlds[lr * L_PAD + k0] : 0.f;
      float x1 = (lr < K_SZ) ? wlds[lr * L_PAD + k0 + 1] : 0.f;
      a[2 * i] = (__bf16)x0;
      a[2 * i + 1] = (__bf16)x1;
    }
    v16bf b0, b1;
#pragma unroll
    for (int i = 0; i < 8; ++i) {
      int l0 = kk * 32 + koff_pair(i, lh);
      b0[2 * i]     = m[l0 * D_SZ + n0 * 16 + lr];
      b0[2 * i + 1] = m[(l0 + 1) * D_SZ + n0 * 16 + lr];
      b1[2 * i]     = m[l0 * D_SZ + n1 * 16 + lr];
      b1[2 * i + 1] = m[(l0 + 1) * D_SZ + n1 * 16 + lr];
    }
    z0 = __builtin_amdgcn_wmma_f32_16x16x32_bf16(false, a, false, b0, (short)0, z0, false, false);
    z1 = __builtin_amdgcn_wmma_f32_16x16x32_bf16(false, a, false, b1, (short)0, z1, false, false);
  }

  // squash: norm[k] = sum_o Z[k][o]^2  (valid C rows live in lanes 0-15)
#pragma unroll
  for (int r = 0; r < 8; ++r) {
    float p = (lh == 0) ? (z0[r] * z0[r] + z1[r] * z1[r]) : 0.f;
#pragma unroll
    for (int s = 16; s > 0; s >>= 1) p += __shfl_xor(p, s, 32);
    if (lane == 0) atomicAdd(&norm[r], p);
  }
  __syncthreads();
  if (tid < K_SZ) {
    float sq = norm[tid];
    scale[tid] = sq / (1.0f + sq) * rsqrtf(sq + 1e-8f);
  }
  __syncthreads();

  if (lh == 0) {
#pragma unroll
    for (int r = 0; r < 8; ++r) {
      float c0 = scale[r] * z0[r];
      float c1 = scale[r] * z1[r];
      caps[r * D_SZ + n0 * 16 + lr] = c0;
      caps[r * D_SZ + n1 * 16 + lr] = c1;
      out[(size_t)b * (K_SZ * D_SZ) + r * D_SZ + n0 * 16 + lr] = c0;
      out[(size_t)b * (K_SZ * D_SZ) + r * D_SZ + n1 * 16 + lr] = c1;
    }
  }
  __syncthreads();

  // delta = caps(16x256, rows>=8 zero) @ mapped^T (256 x L): 13 L-tiles of 16
  v8f d0 = zero, d1 = zero;
  const int t0 = wave, t1 = wave + 8;
  const bool has2 = (t1 < 13);  // wave-uniform
  for (int kk = 0; kk < 8; ++kk) {
    v16bf a;
#pragma unroll
    for (int i = 0; i < 8; ++i) {
      int o0 = kk * 32 + koff_pair(i, lh);
      float x0 = (lr < K_SZ) ? caps[lr * D_SZ + o0] : 0.f;
      float x1 = (lr < K_SZ) ? caps[lr * D_SZ + o0 + 1] : 0.f;
      a[2 * i] = (__bf16)x0;
      a[2 * i + 1] = (__bf16)x1;
    }
    v16bf b0;
#pragma unroll
    for (int i = 0; i < 8; ++i) {
      int o0 = kk * 32 + koff_pair(i, lh);
      int l0 = t0 * 16 + lr;                 // <= 127 < 224
      b0[2 * i]     = m[l0 * D_SZ + o0];
      b0[2 * i + 1] = m[l0 * D_SZ + o0 + 1];
    }
    d0 = __builtin_amdgcn_wmma_f32_16x16x32_bf16(false, a, false, b0, (short)0, d0, false, false);
    if (has2) {
      v16bf b1;
#pragma unroll
      for (int i = 0; i < 8; ++i) {
        int o0 = kk * 32 + koff_pair(i, lh);
        int l1 = t1 * 16 + lr;               // <= 207 < 224
        b1[2 * i]     = m[l1 * D_SZ + o0];
        b1[2 * i + 1] = m[l1 * D_SZ + o0 + 1];
      }
      d1 = __builtin_amdgcn_wmma_f32_16x16x32_bf16(false, a, false, b1, (short)0, d1, false, false);
    }
  }

  if (lh == 0) {
#pragma unroll
    for (int r = 0; r < 8; ++r) {
      int l = t0 * 16 + lr;
      if (l < L_SZ) delta_part[((size_t)b * K_SZ + r) * L_SZ + l] = d0[r];
      if (has2) {
        int l2 = t1 * 16 + lr;
        if (l2 < L_SZ) delta_part[((size_t)b * K_SZ + r) * L_SZ + l2] = d1[r];
      }
    }
  }
}

// ---------------------------------------------------------------------------
// logits[k,l] += sum_b delta_part[b,k,l]; one block per (k,l)
// ---------------------------------------------------------------------------
__global__ void update_logits(const float* __restrict__ delta_part,
                              float* __restrict__ logits) {
  __shared__ float red[256];
  const int idx = blockIdx.x;  // 0..1599 = k*200 + l
  float s = 0.f;
  for (int b = threadIdx.x; b < B_SZ; b += 256)
    s += delta_part[(size_t)b * (K_SZ * L_SZ) + idx];
  red[threadIdx.x] = s;
  __syncthreads();
  for (int off = 128; off > 0; off >>= 1) {
    if (threadIdx.x < off) red[threadIdx.x] += red[threadIdx.x + off];
    __syncthreads();
  }
  if (threadIdx.x == 0) logits[idx] += red[0];
}

extern "C" void kernel_launch(void* const* d_in, const int* in_sizes, int n_in,
                              void* d_out, int out_size, void* d_ws, size_t ws_size,
                              hipStream_t stream) {
  (void)in_sizes; (void)n_in; (void)out_size; (void)ws_size;
  const float* X   = (const float*)d_in[0];
  const int*   seq = (const int*)d_in[1];
  const float* RL  = (const float*)d_in[2];
  const float* W   = (const float*)d_in[3];
  float* out = (float*)d_out;

  unsigned char* ws = (unsigned char*)d_ws;
  unsigned short* mapped   = (unsigned short*)(ws + WS_MAPPED_OFF);
  unsigned*       wfrag    = (unsigned*)(ws + WS_WFRAG_OFF);
  float*          logits   = (float*)(ws + WS_LOGITS_OFF);
  float*          delta    = (float*)(ws + WS_DELTA_OFF);

  prep_wfrag<<<128, 256, 0, stream>>>(W, wfrag);
  init_logits<<<7, 256, 0, stream>>>(RL, logits);
  gemm_map<<<1600, 256, 131072, stream>>>(X, wfrag, mapped);

  const size_t lds2 = (size_t)L_PAD * D_SZ * 2 +
                      (size_t)(K_SZ * L_PAD + K_SZ * D_SZ + 2 * K_SZ) * 4;
  for (int it = 0; it < 3; ++it) {
    routing<<<B_SZ, 256, lds2, stream>>>(mapped, seq, logits, out, delta);
    update_logits<<<K_SZ * L_SZ, 256, 0, stream>>>(delta, logits);
  }
}
